// GSMP_4879082848655
// MI455X (gfx1250) — matrix operand order repaired and checked by reading, not compiled, not run on hardware
//
#include <hip/hip_runtime.h>
#include <hip/hip_bf16.h>
#include <float.h>

#define NNODES 100000
#define NEDGES 1600000

typedef __bf16 bf16x4  __attribute__((ext_vector_type(4)));
typedef __bf16 bf16x8  __attribute__((ext_vector_type(8)));
typedef __bf16 bf16x16 __attribute__((ext_vector_type(16)));
typedef float  f32x8   __attribute__((ext_vector_type(8)));

#define WMMA_BF16(a, b, c) \
  __builtin_amdgcn_wmma_f32_16x16x32_bf16(false, (a), false, (b), (short)0, (c), false, false)

__device__ __forceinline__ void lds_fence() {
  // LDS ops from one wave execute in order; this waits for completion and is a
  // compiler reordering barrier so store->load through LDS is safe intra-wave.
  asm volatile("s_wait_dscnt 0" ::: "memory");
}

// A fragment (16x32 bf16, MxK): lane holds row M=lane%16.
// lanes 0-15: K = kb*32 + {0..15}; lanes 16-31: K = kb*32 + {16..31}
__device__ __forceinline__ bf16x16 load_a_frag(const __bf16* row, int kb, int lane) {
  int off = kb * 32 + ((lane & 16) ? 16 : 0);
  bf16x8 lo = *(const bf16x8*)(row + off);
  bf16x8 hi = *(const bf16x8*)(row + off + 8);
  return __builtin_shufflevector(lo, hi, 0,1,2,3,4,5,6,7,8,9,10,11,12,13,14,15);
}

// B fragment (32x16 bf16, KxN): WT stored transposed [N][K] so each lane reads
// 16 contiguous K values for col N=lane%16 (lanes>=16 read K+16).
__device__ __forceinline__ bf16x16 load_b_frag(const __bf16* WT, int ldk, int nb, int kb, int lane) {
  int n = nb * 16 + (lane & 15);
  int kbase = kb * 32 + ((lane & 16) ? 16 : 0);
  return *(const bf16x16*)(WT + n * ldk + kbase);
}

__device__ __forceinline__ bf16x4 cvt4(float4 v) {
  return (bf16x4){(__bf16)v.x, (__bf16)v.y, (__bf16)v.z, (__bf16)v.w};
}

__global__ __launch_bounds__(256) void init_acc_kernel(float* s1, float* mx2, float* mn3,
                                                       float* s4, float* deg) {
  int i = blockIdx.x * 256 + threadIdx.x;
  if (i < NNODES * 32) {
    s1[i]  = 0.f;
    s4[i]  = 0.f;
    mx2[i] = -FLT_MAX;
    mn3[i] =  FLT_MAX;
  }
  if (i < NNODES) deg[i] = 0.f;
}

__global__ __launch_bounds__(256) void edge_kernel(
    const float* __restrict__ in_vc, const float* __restrict__ in_ve,
    const int* __restrict__ src, const int* __restrict__ dst,
    const float* __restrict__ Wm1, const float* __restrict__ bm1,
    const float* __restrict__ Wm2, const float* __restrict__ bm2,
    const float* __restrict__ We,  const float* __restrict__ be,
    float* __restrict__ out_ve,
    float* __restrict__ s1, float* __restrict__ mx2,
    float* __restrict__ mn3, float* __restrict__ s4, float* __restrict__ deg) {
  __shared__ __attribute__((aligned(32))) __bf16 sW1T[96 * 96];    // [n][k]
  __shared__ __attribute__((aligned(32))) __bf16 sW2T[144 * 96];   // padded 129->144 cols
  __shared__ __attribute__((aligned(32))) __bf16 sWeT[32 * 160];
  __shared__ float sB1[96], sB2[144], sBe[32];
  __shared__ __attribute__((aligned(32))) __bf16 sXH[8 * 16 * 96];   // x then h, per wave
  __shared__ __attribute__((aligned(32))) __bf16 sES[8 * 16 * 160];  // [f(128), ve(32)]
  __shared__ float sK[8 * 16];  // sigmoid gate per edge row
  __shared__ int   sD[8 * 16];  // dst node per edge row

  const int tid = threadIdx.x;
  // --- stage weights transposed as bf16 (amortized across 128 edges) ---
  for (int i = tid; i < 96 * 96; i += 256) {
    int n = i / 96, k = i - n * 96;
    sW1T[i] = (__bf16)Wm1[k * 96 + n];
  }
  for (int i = tid; i < 144 * 96; i += 256) {
    int n = i / 96, k = i - n * 96;
    sW2T[i] = (__bf16)((n < 129) ? Wm2[k * 129 + n] : 0.f);
  }
  for (int i = tid; i < 32 * 160; i += 256) {
    int n = i / 160, k = i - n * 160;
    sWeT[i] = (__bf16)We[k * 32 + n];
  }
  if (tid < 96)  sB1[tid] = bm1[tid];
  if (tid < 144) sB2[tid] = (tid < 129) ? bm2[tid] : 0.f;
  if (tid < 32)  sBe[tid] = be[tid];

  const int wid = tid >> 5, lane = tid & 31;
  const int r = lane & 15, hh = lane >> 4;
  const int ebase = (blockIdx.x * 8 + wid) * 16;
  const int e = ebase + r;
  const int sn = src[e], dn0 = dst[e];

  // --- stage x = [vc[src], vc[dst], ve] as bf16; ve also into es cols 128..159 ---
  {
    __bf16* xr = sXH + (wid * 16 + r) * 96;
    __bf16* er = sES + (wid * 16 + r) * 160;
    const float* ps = in_vc + (size_t)sn * 32 + hh * 16;
    const float* pd = in_vc + (size_t)dn0 * 32 + hh * 16;
    const float* pe = in_ve + (size_t)e * 32 + hh * 16;
#pragma unroll
    for (int i = 0; i < 4; ++i)
      *(bf16x4*)(xr + hh * 16 + i * 4) = cvt4(((const float4*)ps)[i]);
#pragma unroll
    for (int i = 0; i < 4; ++i)
      *(bf16x4*)(xr + 32 + hh * 16 + i * 4) = cvt4(((const float4*)pd)[i]);
#pragma unroll
    for (int i = 0; i < 4; ++i) {
      bf16x4 b = cvt4(((const float4*)pe)[i]);
      *(bf16x4*)(xr + 64 + hh * 16 + i * 4) = b;
      *(bf16x4*)(er + 128 + hh * 16 + i * 4) = b;
    }
    if (hh == 0) {
      sD[wid * 16 + r] = dn0;
      __hip_atomic_fetch_add(deg + dn0, 1.0f, __ATOMIC_RELAXED, __HIP_MEMORY_SCOPE_AGENT);
    }
  }
  __syncthreads();

  const int coln = lane & 15;
  const int off8 = (lane & 16) ? 8 : 0;
  const __bf16* xrow = sXH + (wid * 16 + r) * 96;

  // --- stage 1: h = relu(x @ Wm1 + b1), 16x96 = 6 N-blocks x 3 K-blocks ---
  bf16x16 a0 = load_a_frag(xrow, 0, lane);
  bf16x16 a1 = load_a_frag(xrow, 1, lane);
  bf16x16 a2 = load_a_frag(xrow, 2, lane);
#pragma unroll
  for (int nb = 0; nb < 6; ++nb) {
    f32x8 c = {0.f, 0.f, 0.f, 0.f, 0.f, 0.f, 0.f, 0.f};
    c = WMMA_BF16(a0, load_b_frag(sW1T, 96, nb, 0, lane), c);
    c = WMMA_BF16(a1, load_b_frag(sW1T, 96, nb, 1, lane), c);
    c = WMMA_BF16(a2, load_b_frag(sW1T, 96, nb, 2, lane), c);
    float b = sB1[nb * 16 + coln];
#pragma unroll
    for (int rr = 0; rr < 8; ++rr) {  // D layout: lane holds col N; rows rr+off8
      float v = fmaxf(c[rr] + b, 0.f);
      sXH[(wid * 16 + rr + off8) * 96 + nb * 16 + coln] = (__bf16)v;
    }
  }
  lds_fence();

  // --- stage 2: y = h @ Wm2 + b2 (129 cols -> 9 N-blocks), gate + scatter ---
  bf16x16 h0 = load_a_frag(xrow, 0, lane);
  bf16x16 h1 = load_a_frag(xrow, 1, lane);
  bf16x16 h2 = load_a_frag(xrow, 2, lane);
#pragma unroll
  for (int nb = 0; nb < 9; ++nb) {
    f32x8 c = {0.f, 0.f, 0.f, 0.f, 0.f, 0.f, 0.f, 0.f};
    c = WMMA_BF16(h0, load_b_frag(sW2T, 96, nb, 0, lane), c);
    c = WMMA_BF16(h1, load_b_frag(sW2T, 96, nb, 1, lane), c);
    c = WMMA_BF16(h2, load_b_frag(sW2T, 96, nb, 2, lane), c);
    float b = sB2[nb * 16 + coln];
    if (nb == 0) {  // col 0 is the gate: k = sigmoid(y[:,0])
      if (coln == 0) {
#pragma unroll
        for (int rr = 0; rr < 8; ++rr) {
          float y0 = c[rr] + b;
          sK[wid * 16 + rr + off8] = 1.0f / (1.0f + __expf(-y0));
        }
      }
      lds_fence();
    }
    int fc = nb * 16 + coln - 1;  // f column index (y cols 1..128)
    if (fc >= 0 && fc < 128) {
      int seg = fc >> 5, cc = fc & 31;
#pragma unroll
      for (int rr = 0; rr < 8; ++rr) {
        int M = rr + off8;
        float v = (c[rr] + b) * sK[wid * 16 + M];
        sES[(wid * 16 + M) * 160 + fc] = (__bf16)v;
        int dn = sD[wid * 16 + M];
        size_t off = (size_t)dn * 32 + cc;
        if (seg == 0)
          __hip_atomic_fetch_add(s1 + off, v, __ATOMIC_RELAXED, __HIP_MEMORY_SCOPE_AGENT);
        else if (seg == 1)
          __hip_atomic_fetch_max(mx2 + off, v, __ATOMIC_RELAXED, __HIP_MEMORY_SCOPE_AGENT);
        else if (seg == 2)
          __hip_atomic_fetch_min(mn3 + off, v, __ATOMIC_RELAXED, __HIP_MEMORY_SCOPE_AGENT);
        else
          __hip_atomic_fetch_add(s4 + off, v, __ATOMIC_RELAXED, __HIP_MEMORY_SCOPE_AGENT);
      }
    }
  }
  lds_fence();

  // --- stage 3: out_ve = [f, ve] @ We + be, K=160 = 5 K-blocks, N=32 ---
  const __bf16* erow = sES + (wid * 16 + r) * 160;
  bf16x16 e0 = load_a_frag(erow, 0, lane);
  bf16x16 e1 = load_a_frag(erow, 1, lane);
  bf16x16 e2 = load_a_frag(erow, 2, lane);
  bf16x16 e3 = load_a_frag(erow, 3, lane);
  bf16x16 e4 = load_a_frag(erow, 4, lane);
#pragma unroll
  for (int nb = 0; nb < 2; ++nb) {
    f32x8 c = {0.f, 0.f, 0.f, 0.f, 0.f, 0.f, 0.f, 0.f};
    c = WMMA_BF16(e0, load_b_frag(sWeT, 160, nb, 0, lane), c);
    c = WMMA_BF16(e1, load_b_frag(sWeT, 160, nb, 1, lane), c);
    c = WMMA_BF16(e2, load_b_frag(sWeT, 160, nb, 2, lane), c);
    c = WMMA_BF16(e3, load_b_frag(sWeT, 160, nb, 3, lane), c);
    c = WMMA_BF16(e4, load_b_frag(sWeT, 160, nb, 4, lane), c);
    float b = sBe[nb * 16 + coln];
#pragma unroll
    for (int rr = 0; rr < 8; ++rr)
      out_ve[(size_t)(ebase + rr + off8) * 32 + nb * 16 + coln] = c[rr] + b;
  }
}

__global__ __launch_bounds__(256) void node_kernel(
    const float* __restrict__ in_vc, const float* __restrict__ Wr, const float* __restrict__ br,
    const float* __restrict__ s1, const float* __restrict__ mx2, const float* __restrict__ mn3,
    const float* __restrict__ s4, const float* __restrict__ deg, float* __restrict__ out_vc) {
  __shared__ __attribute__((aligned(32))) __bf16 sWrT[32 * 160];
  __shared__ float sBr[32];
  __shared__ __attribute__((aligned(32))) __bf16 sNS[8 * 16 * 160];

  const int tid = threadIdx.x;
  for (int i = tid; i < 32 * 160; i += 256) {
    int n = i / 160, k = i - n * 160;
    sWrT[i] = (__bf16)Wr[k * 32 + n];
  }
  if (tid < 32) sBr[tid] = br[tid];
  __syncthreads();

  const int wid = tid >> 5, lane = tid & 31;
  const int tile = blockIdx.x * 8 + wid;
  if (tile >= NNODES / 16) return;  // 6250 tiles
  const int r = lane & 15, hh = lane >> 4;
  const int node = tile * 16 + r;

  float dg = deg[node];
  float msk = dg > 0.f ? 1.f : 0.f;
  float inv = 1.f / fmaxf(dg, 1.f);
  __bf16* nr = sNS + (wid * 16 + r) * 160;
  const size_t base = (size_t)node * 32 + hh * 16;
#pragma unroll
  for (int i = 0; i < 4; ++i)
    *(bf16x4*)(nr + hh * 16 + i * 4) = cvt4(((const float4*)(in_vc + base))[i]);
#pragma unroll
  for (int i = 0; i < 4; ++i)
    *(bf16x4*)(nr + 32 + hh * 16 + i * 4) = cvt4(((const float4*)(s1 + base))[i]);
#pragma unroll
  for (int i = 0; i < 4; ++i) {
    float4 v = ((const float4*)(mx2 + base))[i];
    v.x *= msk; v.y *= msk; v.z *= msk; v.w *= msk;  // empty nodes -> 0
    *(bf16x4*)(nr + 64 + hh * 16 + i * 4) = cvt4(v);
  }
#pragma unroll
  for (int i = 0; i < 4; ++i) {
    float4 v = ((const float4*)(mn3 + base))[i];
    v.x *= msk; v.y *= msk; v.z *= msk; v.w *= msk;
    *(bf16x4*)(nr + 96 + hh * 16 + i * 4) = cvt4(v);
  }
#pragma unroll
  for (int i = 0; i < 4; ++i) {
    float4 v = ((const float4*)(s4 + base))[i];
    v.x *= inv; v.y *= inv; v.z *= inv; v.w *= inv;  // mean
    *(bf16x4*)(nr + 128 + hh * 16 + i * 4) = cvt4(v);
  }
  asm volatile("s_wait_dscnt 0" ::: "memory");

  const __bf16* nrow = sNS + (wid * 16 + r) * 160;
  bf16x16 n0 = load_a_frag(nrow, 0, lane);
  bf16x16 n1 = load_a_frag(nrow, 1, lane);
  bf16x16 n2 = load_a_frag(nrow, 2, lane);
  bf16x16 n3 = load_a_frag(nrow, 3, lane);
  bf16x16 n4 = load_a_frag(nrow, 4, lane);
  const int coln = lane & 15;
  const int off8 = (lane & 16) ? 8 : 0;
#pragma unroll
  for (int nb = 0; nb < 2; ++nb) {
    f32x8 c = {0.f, 0.f, 0.f, 0.f, 0.f, 0.f, 0.f, 0.f};
    c = WMMA_BF16(n0, load_b_frag(sWrT, 160, nb, 0, lane), c);
    c = WMMA_BF16(n1, load_b_frag(sWrT, 160, nb, 1, lane), c);
    c = WMMA_BF16(n2, load_b_frag(sWrT, 160, nb, 2, lane), c);
    c = WMMA_BF16(n3, load_b_frag(sWrT, 160, nb, 3, lane), c);
    c = WMMA_BF16(n4, load_b_frag(sWrT, 160, nb, 4, lane), c);
    float b = sBr[nb * 16 + coln];
#pragma unroll
    for (int rr = 0; rr < 8; ++rr)
      out_vc[(size_t)(tile * 16 + rr + off8) * 32 + nb * 16 + coln] = c[rr] + b;
  }
}

extern "C" void kernel_launch(void* const* d_in, const int* in_sizes, int n_in,
                              void* d_out, int out_size, void* d_ws, size_t ws_size,
                              hipStream_t stream) {
  (void)in_sizes; (void)n_in; (void)out_size; (void)ws_size;
  const float* in_vc = (const float*)d_in[0];
  const float* in_ve = (const float*)d_in[1];
  const int*   src   = (const int*)d_in[2];
  const int*   dst   = (const int*)d_in[3];
  const float* Wm1   = (const float*)d_in[4];
  const float* bm1   = (const float*)d_in[5];
  const float* Wm2   = (const float*)d_in[6];
  const float* bm2   = (const float*)d_in[7];
  const float* Wr    = (const float*)d_in[8];
  const float* br    = (const float*)d_in[9];
  const float* We    = (const float*)d_in[10];
  const float* be    = (const float*)d_in[11];

  float* out_vc = (float*)d_out;
  float* out_ve = out_vc + (size_t)NNODES * 32;

  float* ws  = (float*)d_ws;
  float* s1  = ws;
  float* mx2 = s1 + (size_t)NNODES * 32;
  float* mn3 = mx2 + (size_t)NNODES * 32;
  float* s4  = mn3 + (size_t)NNODES * 32;
  float* deg = s4 + (size_t)NNODES * 32;

  init_acc_kernel<<<(NNODES * 32 + 255) / 256, 256, 0, stream>>>(s1, mx2, mn3, s4, deg);
  edge_kernel<<<NEDGES / 128, 256, 0, stream>>>(in_vc, in_ve, src, dst, Wm1, bm1, Wm2, bm2,
                                                We, be, out_ve, s1, mx2, mn3, s4, deg);
  node_kernel<<<(NNODES / 16 + 7) / 8, 256, 0, stream>>>(in_vc, Wr, br, s1, mx2, mn3, s4, deg,
                                                         out_vc);
}
